// GNN_Heuristic_9929964389268
// MI455X (gfx1250) — matrix operand order repaired and checked by reading, not compiled, not run on hardware
//
#include <hip/hip_runtime.h>
#include <math.h>

typedef __attribute__((ext_vector_type(16))) _Float16 v16h;
typedef __attribute__((ext_vector_type(8)))  float    v8f;

#define NN      50000       // nodes
#define NE      1600000     // edges (without self loops)
#define ET      (NE + NN)   // edges + self loops
#define NL      500000      // label edges
#define CH      128         // feature width at every stage
#define NEG_SLOPE 0.2f

// ---- float <-> ordered-uint encoding for atomic max on floats ----
__device__ __forceinline__ unsigned f_ord(float f){
  unsigned u = __float_as_uint(f);
  return (u & 0x80000000u) ? ~u : (u | 0x80000000u);
}
__device__ __forceinline__ float f_unord(unsigned v){
  return (v & 0x80000000u) ? __uint_as_float(v ^ 0x80000000u) : __uint_as_float(~v);
}
__device__ __forceinline__ float lrelu(float v){ return v > 0.f ? v : NEG_SLOPE * v; }

// edge e in [0, ET): real edges then self loops
__device__ __forceinline__ void edge_sd(const int* __restrict__ ei, int e, int& s, int& d){
  if (e < NE){ s = ei[e]; d = ei[NE + e]; }
  else       { s = e - NE; d = s; }
}

// ---------------- conversion helpers ----------------
__global__ void k_f32_to_f16(const float* __restrict__ in, _Float16* __restrict__ out, int n){
  int i = blockIdx.x * blockDim.x + threadIdx.x;
  if (i < n) out[i] = (_Float16)in[i];
}
// W [K=128][N=128] row-major f32 -> Wt [N][K] f16
__global__ void k_transpose_to_f16(const float* __restrict__ w, _Float16* __restrict__ wt){
  int i = blockIdx.x * blockDim.x + threadIdx.x;   // 16384 threads
  int k = i >> 7, n = i & 127;
  wt[n * CH + k] = (_Float16)w[i];
}
__global__ void k_zero(float* __restrict__ p, int n){
  int i = blockIdx.x * blockDim.x + threadIdx.x;
  if (i < n) p[i] = 0.f;
}
__global__ void k_init_attn(unsigned* __restrict__ m, float* __restrict__ s, int n){
  int i = blockIdx.x * blockDim.x + threadIdx.x;
  if (i < n){ m[i] = 0x007FFFFFu; /* ord(-inf) */ s[i] = 0.f; }
}

// ---------------- WMMA GEMM: C[M][128] = A[M][128](f16) x W(via Bt[n][k] f16), f32 acc --------
// one wave per 16x16 tile; block = 256 thr = 8 waves covering all 8 N-tiles of one M-tile row.
// M must be a multiple of 16 (50000 = 3125*16).
__global__ void k_gemm_wmma(const _Float16* __restrict__ A,
                            const _Float16* __restrict__ Bt,
                            float* __restrict__ C, int M){
  const int lane = threadIdx.x & 31;
  const int wv   = threadIdx.x >> 5;          // N tile 0..7
  const int l16  = lane & 15;
  const bool hi  = lane >= 16;
  const int row  = blockIdx.x * 16 + l16;     // A row for this lane
  const int col  = wv * 16 + l16;             // B column / C column
  const _Float16* arow = A  + (size_t)row * CH;
  const _Float16* bcol = Bt + (size_t)col * CH;
  v8f acc = {};
#pragma unroll
  for (int k0 = 0; k0 < CH; k0 += 32){
    v16h a, b;
    const int ak = k0 + (hi ? 8 : 0);         // ISA A layout: {0..7,16..23} / {8..15,24..31}
    const int bk = k0 + (hi ? 8 : 0);         // B layout matches A: {0..7,16..23} / {8..15,24..31}
#pragma unroll
    for (int j = 0; j < 8; ++j){
      a[j]     = arow[ak + j];
      a[8 + j] = arow[ak + 16 + j];
    }
#pragma unroll
    for (int j = 0; j < 8; ++j){ b[j] = bcol[bk + j]; b[8 + j] = bcol[bk + 16 + j]; }
    acc = __builtin_amdgcn_wmma_f32_16x16x32_f16(false, a, false, b, (short)0, acc, false, false);
  }
  const int rbase = blockIdx.x * 16 + (hi ? 8 : 0);   // C layout: VGPR r -> row r (+8 for hi half)
#pragma unroll
  for (int r = 0; r < 8; ++r)
    C[(size_t)(rbase + r) * CH + col] = acc[r];
  (void)M;
}

// ---------------- per-node attention projections: as[n,h]=h_row . a_src[h] ----------------
__global__ void k_rowdot(const float* __restrict__ Hm, const float* __restrict__ a_src,
                         const float* __restrict__ a_dst, float* __restrict__ as,
                         float* __restrict__ ad, int H, int Cc, int total){
  int i = blockIdx.x * blockDim.x + threadIdx.x;
  if (i >= total) return;
  int n = i / H, h = i - n * H;
  const float* row = Hm + (size_t)n * CH + h * Cc;
  const float* ps = a_src + h * Cc;
  const float* pd = a_dst + h * Cc;
  float vs = 0.f, vd = 0.f;
  for (int c = 0; c < Cc; ++c){ float v = row[c]; vs += v * ps[c]; vd += v * pd[c]; }
  as[i] = vs; ad[i] = vd;
}

// ---------------- edge passes ----------------
__global__ void k_edge_max(const int* __restrict__ ei, const float* __restrict__ as,
                           const float* __restrict__ ad, unsigned* __restrict__ m, int H){
  int e = blockIdx.x * blockDim.x + threadIdx.x;
  if (e >= ET) return;
  int s, d; edge_sd(ei, e, s, d);
  for (int h = 0; h < H; ++h){
    float v = lrelu(as[s * H + h] + ad[d * H + h]);
    atomicMax(&m[d * H + h], f_ord(v));
  }
}
__global__ void k_edge_sum(const int* __restrict__ ei, const float* __restrict__ as,
                           const float* __restrict__ ad, const unsigned* __restrict__ m,
                           float* __restrict__ ssum, int H){
  int e = blockIdx.x * blockDim.x + threadIdx.x;
  if (e >= ET) return;
  int s, d; edge_sd(ei, e, s, d);
  for (int h = 0; h < H; ++h){
    float v = lrelu(as[s * H + h] + ad[d * H + h]);
    atomicAdd(&ssum[d * H + h], __expf(v - f_unord(m[d * H + h])));
  }
}
// one wave per edge; lane -> 4 channels; head = channel/ (CH/H)
__global__ void k_edge_scatter(const int* __restrict__ ei, const float* __restrict__ Hm,
                               const float* __restrict__ as, const float* __restrict__ ad,
                               const unsigned* __restrict__ m, const float* __restrict__ ssum,
                               float* __restrict__ out, int H){
  const int lane = threadIdx.x & 31;
  const int e = blockIdx.x * 8 + (threadIdx.x >> 5);
  if (e >= ET) return;
  int s, d; edge_sd(ei, e, s, d);
  const int Cc = CH / H;
  const int c0 = lane * 4;
  const int h  = c0 / Cc;
  float v = lrelu(as[s * H + h] + ad[d * H + h]);
  float alpha = __expf(v - f_unord(m[d * H + h])) / (ssum[d * H + h] + 1e-16f);
  const float* hrow = Hm + (size_t)s * CH;
  float* orow = out + (size_t)d * CH;
#pragma unroll
  for (int j = 0; j < 4; ++j)
    atomicAdd(&orow[c0 + j], hrow[c0 + j] * alpha);
}

// ---------------- bias + ELU + downcast ----------------
__global__ void k_elu_h16(const float* __restrict__ o, const float* __restrict__ b1,
                          _Float16* __restrict__ h16, int n){
  int i = blockIdx.x * blockDim.x + threadIdx.x;
  if (i >= n) return;
  float v = o[i] + b1[i & (CH - 1)];
  v = v > 0.f ? v : (__expf(v) - 1.f);
  h16[i] = (_Float16)v;
}

// ---------------- link prediction head: wave per pair ----------------
__global__ void k_link(const int* __restrict__ eli, const float* __restrict__ z0,
                       const float* __restrict__ b2, const float* __restrict__ wfc,
                       const float* __restrict__ bfc, float* __restrict__ out){
  const int lane = threadIdx.x & 31;
  const int p = blockIdx.x * 8 + (threadIdx.x >> 5);
  if (p >= NL) return;
  const int i0 = eli[p];
  const int i1 = eli[NL + p];
  const int c0 = lane * 4;
  float acc = 0.f;
#pragma unroll
  for (int j = 0; j < 4; ++j){
    int c = c0 + j;
    float bb = b2[c];
    acc += (z0[(size_t)i0 * CH + c] + bb) * wfc[c];
    acc += (z0[(size_t)i1 * CH + c] + bb) * wfc[CH + c];
  }
#pragma unroll
  for (int o = 16; o > 0; o >>= 1) acc += __shfl_xor(acc, o, 32);
  if (lane == 0){
    float v = acc + bfc[0];
    out[p] = 1.f / (1.f + __expf(-v));
  }
}

extern "C" void kernel_launch(void* const* d_in, const int* in_sizes, int n_in,
                              void* d_out, int out_size, void* d_ws, size_t ws_size,
                              hipStream_t stream){
  (void)in_sizes; (void)n_in; (void)out_size; (void)ws_size;
  const float* x    = (const float*)d_in[0];
  const int*   ei   = (const int*)d_in[1];
  const int*   eli  = (const int*)d_in[2];
  const float* W1   = (const float*)d_in[3];
  const float* as1w = (const float*)d_in[4];
  const float* ad1w = (const float*)d_in[5];
  const float* b1   = (const float*)d_in[6];
  const float* W2   = (const float*)d_in[7];
  const float* as2w = (const float*)d_in[8];
  const float* ad2w = (const float*)d_in[9];
  const float* b2   = (const float*)d_in[10];
  const float* Wfc  = (const float*)d_in[11];
  const float* bfc  = (const float*)d_in[12];
  float* out = (float*)d_out;

  char* ws = (char*)d_ws;
  size_t off = 0;
  auto alloc = [&](size_t bytes)->char*{
    char* p = ws + off;
    off += (bytes + 255) & ~(size_t)255;
    return p;
  };
  _Float16* x16 = (_Float16*)alloc((size_t)NN * CH * 2);
  _Float16* w1t = (_Float16*)alloc(CH * CH * 2);
  _Float16* w2t = (_Float16*)alloc(CH * CH * 2);
  float*    Hbuf = (float*)alloc((size_t)NN * CH * 4);     // H1, then reused as Z0
  float*    as1  = (float*)alloc((size_t)NN * 4 * 4);
  float*    ad1  = (float*)alloc((size_t)NN * 4 * 4);
  unsigned* m1   = (unsigned*)alloc((size_t)NN * 4 * 4);
  float*    s1   = (float*)alloc((size_t)NN * 4 * 4);
  float*    obuf = (float*)alloc((size_t)NN * CH * 4);     // out1, then reused as out2
  _Float16* h16  = (_Float16*)alloc((size_t)NN * CH * 2);
  float*    as2  = (float*)alloc((size_t)NN * 4);
  float*    ad2  = (float*)alloc((size_t)NN * 4);
  unsigned* m2   = (unsigned*)alloc((size_t)NN * 4);
  float*    s2   = (float*)alloc((size_t)NN * 4);

  const int B = 256;
  const int NC = NN * CH;  // 6,400,000

  // ---------- layer 1 ----------
  k_f32_to_f16     <<<(NC + B - 1) / B, B, 0, stream>>>(x, x16, NC);
  k_transpose_to_f16<<<(CH * CH + B - 1) / B, B, 0, stream>>>(W1, w1t);
  k_transpose_to_f16<<<(CH * CH + B - 1) / B, B, 0, stream>>>(W2, w2t);
  k_gemm_wmma      <<<NN / 16, 256, 0, stream>>>(x16, w1t, Hbuf, NN);         // H1 = x@W1
  k_rowdot         <<<(NN * 4 + B - 1) / B, B, 0, stream>>>(Hbuf, as1w, ad1w, as1, ad1, 4, 32, NN * 4);
  k_init_attn      <<<(NN * 4 + B - 1) / B, B, 0, stream>>>(m1, s1, NN * 4);
  k_zero           <<<(NC + B - 1) / B, B, 0, stream>>>(obuf, NC);
  k_edge_max       <<<(ET + B - 1) / B, B, 0, stream>>>(ei, as1, ad1, m1, 4);
  k_edge_sum       <<<(ET + B - 1) / B, B, 0, stream>>>(ei, as1, ad1, m1, s1, 4);
  k_edge_scatter   <<<(ET + 7) / 8, 256, 0, stream>>>(ei, Hbuf, as1, ad1, m1, s1, obuf, 4);
  k_elu_h16        <<<(NC + B - 1) / B, B, 0, stream>>>(obuf, b1, h16, NC);   // h = elu(out1+b1)

  // ---------- layer 2 ----------
  k_gemm_wmma      <<<NN / 16, 256, 0, stream>>>(h16, w2t, Hbuf, NN);         // Z0 = h@W2
  k_rowdot         <<<(NN + B - 1) / B, B, 0, stream>>>(Hbuf, as2w, ad2w, as2, ad2, 1, 128, NN);
  k_init_attn      <<<(NN + B - 1) / B, B, 0, stream>>>(m2, s2, NN);
  k_zero           <<<(NC + B - 1) / B, B, 0, stream>>>(obuf, NC);
  k_edge_max       <<<(ET + B - 1) / B, B, 0, stream>>>(ei, as2, ad2, m2, 1);
  k_edge_sum       <<<(ET + B - 1) / B, B, 0, stream>>>(ei, as2, ad2, m2, s2, 1);
  k_edge_scatter   <<<(ET + 7) / 8, 256, 0, stream>>>(ei, Hbuf, as2, ad2, m2, s2, obuf, 1);

  // ---------- link head: sigmoid(concat(z[i],z[j]) @ Wfc + bfc), z = out2 + b2 ----------
  k_link           <<<(NL + 7) / 8, 256, 0, stream>>>(eli, obuf, b2, Wfc, bfc, out);
}
